// Attention_16595753632059
// MI455X (gfx1250) — matrix operand order repaired and checked by reading, not compiled, not run on hardware
//
#include <hip/hip_runtime.h>

// CDNA5 / gfx1250 fused attention:
//   scores = (Q K^T)/sqrt(S); P = softmax(scores); O = P V
// Outputs: O [B,S,D] then P [B,S,S], concatenated in d_out.
//
// Fast path (needs 50.3MB workspace):
//   1) pack Q,K -> f16 row-major; transpose+pack V -> Vt[B,D,S] f16   (one-time,
//      ~50MB extra HBM write; operands then stream from the 192MB L2)
//   2) fused kernel: all WMMA operands are contiguous 16B/32B f16 chunks ->
//      pure global_load_b128 / ds_load_b128 fragment loads, zero cvt in the
//      hot loops, v_wmma_f32_16x16x32_f16 everywhere. Scores kept as f16 in
//      66KB of LDS (vs 131KB f32) for 4-blocks/WGP occupancy headroom.
// Fallback path (tiny workspace): round-1 kernel converting fp32 on the fly.

typedef __attribute__((ext_vector_type(16))) _Float16 v16h;
typedef __attribute__((ext_vector_type(8)))  _Float16 v8h;
typedef __attribute__((ext_vector_type(8)))  float    v8f;
typedef __attribute__((ext_vector_type(4)))  float    v4f;

#define BATCH 8
#define S_LEN 2048
#define D_DIM 512
#define PH_ROW (S_LEN + 8)   // f16 LDS row stride (halves): 4112B rows, b128-aligned,
                             // lanes land on distinct bank quads
#define SROW   (S_LEN + 4)   // f32 fallback LDS row stride

// ---------------------------------------------------------------------------
// Prepack: fp32 -> f16 elementwise (Q and K)
// ---------------------------------------------------------------------------
__global__ __launch_bounds__(256) void pack_f16_kernel(
    const float* __restrict__ src, _Float16* __restrict__ dst, size_t n)
{
    size_t i = ((size_t)blockIdx.x * blockDim.x + threadIdx.x) * 8;
    const size_t stride = (size_t)gridDim.x * blockDim.x * 8;
    for (; i < n; i += stride) {
        v4f x0 = *(const v4f*)(src + i);
        v4f x1 = *(const v4f*)(src + i + 4);
        v8h h;
        #pragma unroll
        for (int j = 0; j < 4; ++j) {
            h[j]     = (_Float16)x0[j];
            h[4 + j] = (_Float16)x1[j];
        }
        *(v8h*)(dst + i) = h;   // global_store_b128
    }
}

// ---------------------------------------------------------------------------
// Prepack: V[B,S,D] fp32 -> Vt[B,D,S] f16 (64x64 LDS tile transpose)
// ---------------------------------------------------------------------------
__global__ __launch_bounds__(256) void transpose_v_f16_kernel(
    const float* __restrict__ V, _Float16* __restrict__ Vt)
{
    __shared__ _Float16 tile[64 * 72];          // [d][s], padded rows

    const int tilesPer = (S_LEN / 64) * (D_DIM / 64);
    const int b   = blockIdx.x / tilesPer;
    const int r   = blockIdx.x % tilesPer;
    const int s0  = (r / (D_DIM / 64)) * 64;
    const int d0  = (r % (D_DIM / 64)) * 64;

    const int t  = threadIdx.x;
    const int tx = t & 15;                      // 16 x 4-col groups
    const int ty = t >> 4;                      // 16 rows

    const float* vb = V + (size_t)b * S_LEN * D_DIM;
    #pragma unroll
    for (int rr = 0; rr < 4; ++rr) {
        const int srow = ty + rr * 16;
        v4f x = *(const v4f*)(vb + (size_t)(s0 + srow) * D_DIM + d0 + tx * 4);
        #pragma unroll
        for (int i = 0; i < 4; ++i)
            tile[(tx * 4 + i) * 72 + srow] = (_Float16)x[i];
    }
    __syncthreads();

    const int dd = t >> 2;                      // 64 output rows (d)
    const int ch = t & 3;                       // 4 x 16-half chunks (s)
    _Float16* orow = Vt + ((size_t)b * D_DIM + d0 + dd) * (size_t)S_LEN + s0 + ch * 16;
    v8h h0, h1;
    #pragma unroll
    for (int i = 0; i < 8; ++i) {
        h0[i] = tile[dd * 72 + ch * 16 + i];
        h1[i] = tile[dd * 72 + ch * 16 + 8 + i];
    }
    *(v8h*)orow       = h0;
    *(v8h*)(orow + 8) = h1;
}

// ---------------------------------------------------------------------------
// Fast fused attention: f16 operands, f32 accumulate
// ---------------------------------------------------------------------------
__global__ __launch_bounds__(256) void attn_f16_kernel(
    const _Float16* __restrict__ Qh,   // [B,S,D] f16
    const _Float16* __restrict__ Kh,   // [B,S,D] f16
    const _Float16* __restrict__ Vt,   // [B,D,S] f16 (transposed)
    float* __restrict__ Out,           // [B,S,D]
    float* __restrict__ W)             // [B,S,S]
{
    extern __shared__ _Float16 ph[];   // 16 x PH_ROW halves: scores -> P

    const int tid  = threadIdx.x;
    const int lane = tid & 31;         // wave32
    const int wave = tid >> 5;         // 0..7
    const int hi   = lane >> 4;
    const int lm   = lane & 15;

    const int b   = blockIdx.x >> 7;
    const int qt  = blockIdx.x & 127;
    const int qm0 = qt << 4;

    const float rscale = 0.022097086912079608f;   // 1/sqrt(2048)

    // ---- Phase 1: S = Q K^T ------------------------------------------------
    // A (16x32 f16): lane m=lm; halves j<8 -> k=hi*8+j ; j>=8 -> k=16+hi*8+(j-8)
    v16h a[16];
    {
        const _Float16* qrow = Qh + ((size_t)b * S_LEN + qm0 + lm) * (size_t)D_DIM;
        #pragma unroll
        for (int f = 0; f < 16; ++f) {
            v8h lo = *(const v8h*)(qrow + f * 32 + hi * 8);        // b128
            v8h hh = *(const v8h*)(qrow + f * 32 + 16 + hi * 8);   // b128
            #pragma unroll
            for (int j = 0; j < 8; ++j) { a[f][j] = lo[j]; a[f][8 + j] = hh[j]; }
        }
    }

    for (int kt = wave; kt < (S_LEN / 16); kt += 8) {
        const int kn0 = kt << 4;
        // B = K^T (32x16 f16): lane n=lm; halves j<8 -> k=hi*8+j ; j>=8 -> k=16+hi*8+(j-8)
        const _Float16* krow = Kh + ((size_t)b * S_LEN + kn0 + lm) * (size_t)D_DIM;
        v8f acc = {};
        #pragma unroll
        for (int f = 0; f < 16; ++f) {
            v8h b0 = *(const v8h*)(krow + f * 32 + hi * 8);
            v8h b1 = *(const v8h*)(krow + f * 32 + 16 + hi * 8);
            v16h bf;
            #pragma unroll
            for (int j = 0; j < 8; ++j) { bf[j] = b0[j]; bf[8 + j] = b1[j]; }
            acc = __builtin_amdgcn_wmma_f32_16x16x32_f16(
                false, a[f], false, bf, (short)0, acc, false, false);
        }
        #pragma unroll
        for (int i = 0; i < 8; ++i)
            ph[(i + 8 * hi) * PH_ROW + kn0 + lm] = (_Float16)(acc[i] * rscale);
    }
    __syncthreads();

    // ---- Phase 2: softmax rows; write f32 weights; keep f16 P in LDS -------
    {
        const int r   = tid >> 4;
        const int sub = tid & 15;
        _Float16* row = ph + r * PH_ROW;

        float m = -3.4e38f;
        for (int j = sub; j < S_LEN; j += 16) m = fmaxf(m, (float)row[j]);
        #pragma unroll
        for (int off = 8; off; off >>= 1) m = fmaxf(m, __shfl_xor(m, off, 16));

        float s = 0.f;
        for (int j = sub; j < S_LEN; j += 16) {
            float e = __expf((float)row[j] - m);
            row[j] = (_Float16)e;
            s += e;
        }
        #pragma unroll
        for (int off = 8; off; off >>= 1) s += __shfl_xor(s, off, 16);
        const float inv = 1.f / s;

        float* wrow = W + ((size_t)b * S_LEN + qm0 + r) * (size_t)S_LEN;
        for (int j = sub; j < S_LEN; j += 16) {
            float p = (float)row[j] * inv;
            row[j]  = (_Float16)p;
            wrow[j] = p;                       // coalesced f32 weights out
        }
    }
    __syncthreads();

    // ---- Phase 3: O = P V --------------------------------------------------
    {
        const int n0 = wave * 64;
        v8f acc[4] = {{}, {}, {}, {}};

        for (int kk = 0; kk < S_LEN; kk += 32) {
            // A from P in LDS: two ds_load_b128 per fragment
            const _Float16* prow = ph + lm * PH_ROW + kk;
            v8h l0 = *(const v8h*)(prow + hi * 16);
            v8h l1 = *(const v8h*)(prow + hi * 16 + 8);
            v16h af;
            #pragma unroll
            for (int j = 0; j < 8; ++j) { af[j] = l0[j]; af[8 + j] = l1[j]; }

            // B from Vt rows: two global_load_b128 per fragment (contiguous!)
            #pragma unroll
            for (int t = 0; t < 4; ++t) {
                const _Float16* vrow =
                    Vt + ((size_t)b * D_DIM + n0 + t * 16 + lm) * (size_t)S_LEN + kk + hi * 16;
                v8h b0 = *(const v8h*)(vrow);
                v8h b1 = *(const v8h*)(vrow + 8);
                v16h bf;
                #pragma unroll
                for (int j = 0; j < 8; ++j) { bf[j] = b0[j]; bf[8 + j] = b1[j]; }
                acc[t] = __builtin_amdgcn_wmma_f32_16x16x32_f16(
                    false, af, false, bf, (short)0, acc[t], false, false);
            }
        }

        #pragma unroll
        for (int t = 0; t < 4; ++t) {
            const int col = n0 + t * 16 + lm;
            float* orow = Out + ((size_t)b * S_LEN + qm0 + 8 * hi) * (size_t)D_DIM + col;
            #pragma unroll
            for (int i = 0; i < 8; ++i)
                orow[(size_t)i * D_DIM] = acc[t][i];
        }
    }
}

// ---------------------------------------------------------------------------
// Fallback (tiny workspace): fp32 loads with on-the-fly f16 conversion
// ---------------------------------------------------------------------------
__global__ __launch_bounds__(256) void attn_fused_f32_kernel(
    const float* __restrict__ Q, const float* __restrict__ K,
    const float* __restrict__ V, float* __restrict__ Out, float* __restrict__ W)
{
    extern __shared__ float sc[];

    const int tid  = threadIdx.x;
    const int lane = tid & 31;
    const int wave = tid >> 5;
    const int hi   = lane >> 4;
    const int lm   = lane & 15;

    const int b   = blockIdx.x >> 7;
    const int qt  = blockIdx.x & 127;
    const int qm0 = qt << 4;

    const float rscale = 0.022097086912079608f;

    v16h a[16];
    {
        const float* qrow = Q + ((size_t)b * S_LEN + qm0 + lm) * (size_t)D_DIM;
        #pragma unroll
        for (int f = 0; f < 16; ++f) {
            const int base = f * 32 + hi * 8;
            #pragma unroll
            for (int j = 0; j < 8; ++j) {
                a[f][j]     = (_Float16)qrow[base + j];
                a[f][8 + j] = (_Float16)qrow[base + 16 + j];
            }
        }
    }

    for (int kt = wave; kt < (S_LEN / 16); kt += 8) {
        const int kn0 = kt << 4;
        const float* krow = K + ((size_t)b * S_LEN + kn0 + lm) * (size_t)D_DIM;
        v8f acc = {};
        #pragma unroll
        for (int f = 0; f < 16; ++f) {
            const int base = f * 32 + hi * 8;
            v16h bf;
            #pragma unroll
            for (int j = 0; j < 16; ++j) bf[j] = (_Float16)krow[base + (j & 7) + (j >> 3) * 16];
            acc = __builtin_amdgcn_wmma_f32_16x16x32_f16(
                false, a[f], false, bf, (short)0, acc, false, false);
        }
        #pragma unroll
        for (int i = 0; i < 8; ++i)
            sc[(i + 8 * hi) * SROW + kn0 + lm] = acc[i] * rscale;
    }
    __syncthreads();

    {
        const int r   = tid >> 4;
        const int sub = tid & 15;
        float* row = sc + r * SROW;

        float m = -3.4e38f;
        for (int j = sub; j < S_LEN; j += 16) m = fmaxf(m, row[j]);
        #pragma unroll
        for (int off = 8; off; off >>= 1) m = fmaxf(m, __shfl_xor(m, off, 16));

        float s = 0.f;
        for (int j = sub; j < S_LEN; j += 16) { float e = __expf(row[j] - m); row[j] = e; s += e; }
        #pragma unroll
        for (int off = 8; off; off >>= 1) s += __shfl_xor(s, off, 16);
        const float inv = 1.f / s;

        float* wrow = W + ((size_t)b * S_LEN + qm0 + r) * (size_t)S_LEN;
        for (int j = sub; j < S_LEN; j += 16) { float p = row[j] * inv; row[j] = p; wrow[j] = p; }
    }
    __syncthreads();

    {
        const int n0 = wave * 64;
        v8f acc[4] = {{}, {}, {}, {}};

        for (int kk = 0; kk < S_LEN; kk += 32) {
            v16h af;
            #pragma unroll
            for (int j = 0; j < 8; ++j) {
                af[j]     = (_Float16)sc[lm * SROW + kk + hi * 16 + j];
                af[8 + j] = (_Float16)sc[lm * SROW + kk + hi * 16 + 8 + j];
            }
            const float* vbase = V + ((size_t)b * S_LEN + kk + hi * 16) * (size_t)D_DIM;
            #pragma unroll
            for (int t = 0; t < 4; ++t) {
                const int col = n0 + t * 16 + lm;
                v16h bf;
                #pragma unroll
                for (int j = 0; j < 16; ++j)
                    bf[j] = (_Float16)vbase[(size_t)j * D_DIM + col];
                acc[t] = __builtin_amdgcn_wmma_f32_16x16x32_f16(
                    false, af, false, bf, (short)0, acc[t], false, false);
            }
        }

        #pragma unroll
        for (int t = 0; t < 4; ++t) {
            const int col = n0 + t * 16 + lm;
            float* orow = Out + ((size_t)b * S_LEN + qm0 + 8 * hi) * (size_t)D_DIM + col;
            #pragma unroll
            for (int i = 0; i < 8; ++i)
                orow[(size_t)i * D_DIM] = acc[t][i];
        }
    }
}

// ---------------------------------------------------------------------------
extern "C" void kernel_launch(void* const* d_in, const int* in_sizes, int n_in,
                              void* d_out, int out_size, void* d_ws, size_t ws_size,
                              hipStream_t stream) {
    (void)in_sizes; (void)n_in; (void)out_size;
    const float* q = (const float*)d_in[0];
    const float* k = (const float*)d_in[1];
    const float* v = (const float*)d_in[2];
    float* out = (float*)d_out;                              // [B,S,D]
    float* w   = out + (size_t)BATCH * S_LEN * D_DIM;        // [B,S,S]

    const size_t elems = (size_t)BATCH * S_LEN * D_DIM;      // 8,388,608
    const size_t need  = 3 * elems * sizeof(_Float16);       // 50.3 MB

    const dim3 grid(BATCH * (S_LEN / 16));                   // 1024
    const dim3 block(256);                                   // 8 wave32s

    if (ws_size >= need) {
        _Float16* qh = (_Float16*)d_ws;
        _Float16* kh = qh + elems;
        _Float16* vt = kh + elems;
        pack_f16_kernel<<<4096, 256, 0, stream>>>(q, qh, elems);
        pack_f16_kernel<<<4096, 256, 0, stream>>>(k, kh, elems);
        transpose_v_f16_kernel<<<BATCH * (S_LEN / 64) * (D_DIM / 64), 256, 0, stream>>>(v, vt);
        attn_f16_kernel<<<grid, block, (size_t)16 * PH_ROW * sizeof(_Float16), stream>>>(
            qh, kh, vt, out, w);
    } else {
        attn_fused_f32_kernel<<<grid, block, (size_t)16 * SROW * sizeof(float), stream>>>(
            q, k, v, out, w);
    }
}
